// CCM_Model_29523605192710
// MI455X (gfx1250) — matrix operand order repaired and checked by reading, not compiled, not run on hardware
//
#include <hip/hip_runtime.h>
#include <cstddef>

// ---- problem constants (match reference) ----
#define CB    8      // batch
#define CS    128    // seq positions
#define CM    32     // triples per position
#define CE    100    // entity dim
#define CR    100    // relation dim
#define CW    300    // word dim
#define CH    512    // hidden
#define CVATT 512
#define CV    40000  // vocab
#define CT    50     // decode steps
#define CTRIP 300    // R + 2E
#define CDECIN 1612
#define CENCIN 500   // W + 2E
#define C3H   1536

typedef __attribute__((ext_vector_type(16))) _Float16 v16h;
typedef __attribute__((ext_vector_type(8)))  float    v8f;

// Branchless masked float4 load: unconditional b128 from a clamped address,
// then v_cndmask the elements. Never generates exec-masked branchy loads.
__device__ __forceinline__ float4 ld4m(const float4* __restrict__ p,
                                       long idx, bool ok)
{
    float4 v = p[ok ? idx : 0L];
    v.x = ok ? v.x : 0.0f;
    v.y = ok ? v.y : 0.0f;
    v.z = ok ? v.z : 0.0f;
    v.w = ok ? v.w : 0.0f;
    return v;
}

// =====================================================================
// Weight prepack (once per launch): f32 weight -> f16 [N, KP] K-major,
// KP = K rounded up to 32, zero padded.  trans=1: src is [K,N] row-major
// (X@W projection weights); trans=0: src is [N,K] (torch GRUCell [out,in]).
// =====================================================================
__global__ void pack_b_kernel(const float* __restrict__ src,
                              _Float16* __restrict__ dst,
                              int N, int K, int KP, int trans)
{
    long i = (long)blockIdx.x * blockDim.x + threadIdx.x;
    if (i >= (long)N * KP) return;
    int n = (int)(i / KP), k = (int)(i % KP);
    float v = 0.0f;
    if (k < K) v = trans ? src[(size_t)k * N + n] : src[(size_t)n * K + k];
    dst[i] = (_Float16)v;
}

// =====================================================================
// WMMA GEMM:  C[M,N] = A[M,K] @ B^T with B prepacked f16 [N, KP].
// REQUIRES K % 4 == 0, KP % 32 == 0, 16B-aligned A (true for every call).
// Block = 128 threads = 4 waves. Wave owns a 16x64 C strip: one A fragment
// shared across 4 N-tiles -> 4 back-to-back v_wmma per k-step. All loads for
// a k-step are issued before any consumption (A: 4 x b128 f32 + cvt;
// B: 4 x v16h = 8 x b128 f16, no conversion, zero padding handles K edge).
// A and B fragments both use contiguous k = 16*(lane/16) + s; D uses the ISA accumulator map.
// =====================================================================
__global__ __launch_bounds__(128) void wmma_gemm_kernel(
    const float* __restrict__ A, const _Float16* __restrict__ Bh,
    float* __restrict__ C, int M, int N, int K, int KP)
{
    const int lane = threadIdx.x & 31;
    const int wave = threadIdx.x >> 5;
    const int n0   = blockIdx.x * 64;                 // 4 N-tiles of 16
    const int m0   = (blockIdx.y * 4 + wave) * 16;
    if (m0 >= M) return;

    const int l16   = lane & 15;
    const int hi    = lane >> 4;
    const int arow  = m0 + l16;          // A: lane%16 -> M row
    const bool aok  = (arow < M);
    const int ahalf = hi * 16;           // A K sub-offset per lane group
    const int bhalf = hi * 16;           // B K sub-offset per lane group
    const int Kc    = K >> 2;            // A float4 chunks per row
    const int KPc   = KP >> 4;           // B v16h chunks per row
    const float4* __restrict__ A4  = (const float4*)A;
    const v16h*   __restrict__ B16 = (const v16h*)Bh;
    const long arow4 = (long)arow * Kc;

    // B row chunk bases (clamped; out-of-range cols never stored)
    long brow[4];
#pragma unroll
    for (int nt = 0; nt < 4; ++nt) {
        const int bcol = n0 + nt * 16 + l16;
        brow[nt] = (bcol < N) ? (long)bcol * KPc : 0L;
    }

    v8f acc[4] = {{}, {}, {}, {}};

    for (int k0 = 0; k0 < KP; k0 += 32) {
        // ---- issue ALL loads for this k-step first ----
        const int kA0 = k0 + ahalf;          // A elements j = 0..7
        const int kA1 = kA0 + 8;             // A elements j = 8..15
        float4 a0 = ld4m(A4, arow4 + ((kA0    ) >> 2), aok && (kA0     < K));
        float4 a1 = ld4m(A4, arow4 + ((kA0 + 4) >> 2), aok && (kA0 + 4 < K));
        float4 a2 = ld4m(A4, arow4 + ((kA1    ) >> 2), aok && (kA1     < K));
        float4 a3 = ld4m(A4, arow4 + ((kA1 + 4) >> 2), aok && (kA1 + 4 < K));
        const long bchunk = (long)((k0 + bhalf) >> 4);
        v16h bf0 = B16[brow[0] + bchunk];
        v16h bf1 = B16[brow[1] + bchunk];
        v16h bf2 = B16[brow[2] + bchunk];
        v16h bf3 = B16[brow[3] + bchunk];

        // ---- convert A fragment (16x32 f16 layout) ----
        v16h af;
        af[0]  = (_Float16)a0.x; af[1]  = (_Float16)a0.y;
        af[2]  = (_Float16)a0.z; af[3]  = (_Float16)a0.w;
        af[4]  = (_Float16)a1.x; af[5]  = (_Float16)a1.y;
        af[6]  = (_Float16)a1.z; af[7]  = (_Float16)a1.w;
        af[8]  = (_Float16)a2.x; af[9]  = (_Float16)a2.y;
        af[10] = (_Float16)a2.z; af[11] = (_Float16)a2.w;
        af[12] = (_Float16)a3.x; af[13] = (_Float16)a3.y;
        af[14] = (_Float16)a3.z; af[15] = (_Float16)a3.w;

        // ---- 4 back-to-back WMMAs on the XDL pipe ----
        acc[0] = __builtin_amdgcn_wmma_f32_16x16x32_f16(
            false, af, false, bf0, (short)0, acc[0], false, false);
        acc[1] = __builtin_amdgcn_wmma_f32_16x16x32_f16(
            false, af, false, bf1, (short)0, acc[1], false, false);
        acc[2] = __builtin_amdgcn_wmma_f32_16x16x32_f16(
            false, af, false, bf2, (short)0, acc[2], false, false);
        acc[3] = __builtin_amdgcn_wmma_f32_16x16x32_f16(
            false, af, false, bf3, (short)0, acc[3], false, false);
    }
    // D f32 16x16: VGPR r -> row m0 + r + (lane>=16 ? 8 : 0); col = bcol
#pragma unroll
    for (int nt = 0; nt < 4; ++nt) {
        const int bcol = n0 + nt * 16 + l16;
        if (bcol < N) {
#pragma unroll
            for (int r = 0; r < 8; ++r) {
                const int m = m0 + r + hi * 8;
                if (m < M) C[(size_t)m * N + bcol] = acc[nt][r];
            }
        }
    }
}

// =====================================================================
// Graph encoder attention fuse: beta = sum_g rh*tanh(hh+th); softmax over
// M=32 triples; g = sum_m a[m]*cat(head,tail).  One block per (b,s).
// g stored in SB layout rows (s*CB + b) for the encoder scan.
// =====================================================================
__global__ __launch_bounds__(128) void graph_fuse_kernel(
    const float* __restrict__ HH, const float* __restrict__ TT,
    const float* __restrict__ RR, const float* __restrict__ head,
    const float* __restrict__ tail, float* __restrict__ g)
{
    const int pos = blockIdx.x;          // b*CS + s
    const int b = pos / CS, s = pos % CS;
    const int t = threadIdx.x;
    __shared__ float part[128];
    __shared__ float beta[CM];
    __shared__ float a[CM];

    const int m = t >> 2, q = t & 3;     // 32 triples x 4 partials
    {
        const float* hh = HH + (size_t)(pos * CM + m) * CE;
        const float* tt = TT + (size_t)(pos * CM + m) * CE;
        const float* rr = RR + (size_t)(pos * CM + m) * CE;
        float acc = 0.0f;
        for (int d = q * 25; d < q * 25 + 25; ++d)
            acc += rr[d] * tanhf(hh[d] + tt[d]);
        part[t] = acc;
    }
    __syncthreads();
    if (q == 0) beta[m] = part[t] + part[t + 1] + part[t + 2] + part[t + 3];
    __syncthreads();
    if (t == 0) {
        float mx = -3.4e38f;
        for (int i = 0; i < CM; ++i) mx = fmaxf(mx, beta[i]);
        float sm = 0.0f;
        for (int i = 0; i < CM; ++i) { float e = __expf(beta[i] - mx); a[i] = e; sm += e; }
        float inv = 1.0f / sm;
        for (int i = 0; i < CM; ++i) a[i] *= inv;
    }
    __syncthreads();
    const int row_out = s * CB + b;
    for (int d = t; d < 2 * CE; d += 128) {
        float acc = 0.0f;
        if (d < CE)
            for (int m2 = 0; m2 < CM; ++m2)
                acc += a[m2] * head[((size_t)pos * CM + m2) * CE + d];
        else
            for (int m2 = 0; m2 < CM; ++m2)
                acc += a[m2] * tail[((size_t)pos * CM + m2) * CE + (d - CE)];
        g[(size_t)row_out * (2 * CE) + d] = acc;
    }
}

// enc_in = cat(word_emb, g) in SB layout
__global__ __launch_bounds__(128) void build_enc_in_kernel(
    const float* __restrict__ wordE, const float* __restrict__ g,
    float* __restrict__ X)
{
    const int pos = blockIdx.x;          // s*CB + b
    const int s = pos / CB, b = pos % CB;
    for (int j = threadIdx.x; j < CENCIN; j += 128) {
        float v = (j < CW) ? wordE[((size_t)b * CS + s) * CW + j]
                           : g[(size_t)pos * (2 * CE) + (j - CW)];
        X[(size_t)pos * CENCIN + j] = v;
    }
}

// all_emb = cat(head, rel, tail) per triple  [B*S*M, 300]
__global__ void build_all_emb_kernel(
    const float* __restrict__ head, const float* __restrict__ rel,
    const float* __restrict__ tail, float* __restrict__ out)
{
    size_t i = (size_t)blockIdx.x * blockDim.x + threadIdx.x;
    const size_t total = (size_t)CB * CS * CM * CTRIP;
    if (i >= total) return;
    size_t row = i / CTRIP; int j = (int)(i % CTRIP);
    float v;
    if (j < CE)            v = head[row * CE + j];
    else if (j < 2 * CE)   v = rel [row * CR + (j - CE)];
    else                   v = tail[row * CE + (j - 2 * CE)];
    out[i] = v;
}

// GRU gate fusion: h_new = (1-z)*n + z*h ; optionally streams y = h_new.
__global__ __launch_bounds__(256) void gru_combine_kernel(
    const float* __restrict__ gi, const float* __restrict__ gh,
    const float* __restrict__ bih, const float* __restrict__ bhh,
    float* __restrict__ h, float* __restrict__ y)
{
    const int b = blockIdx.x;
    for (int j = threadIdx.x; j < CH; j += 256) {
        float ir = gi[b * C3H + j]            + bih[j];
        float iz = gi[b * C3H + CH + j]       + bih[CH + j];
        float in = gi[b * C3H + 2 * CH + j]   + bih[2 * CH + j];
        float hr = gh[b * C3H + j]            + bhh[j];
        float hz = gh[b * C3H + CH + j]       + bhh[CH + j];
        float hn = gh[b * C3H + 2 * CH + j]   + bhh[2 * CH + j];
        float r = 1.0f / (1.0f + __expf(-(ir + hr)));
        float z = 1.0f / (1.0f + __expf(-(iz + hz)));
        float n = tanhf(in + r * hn);
        float hv = h[b * CH + j];
        float nh = (1.0f - z) * n + z * hv;
        h[b * CH + j] = nh;
        if (y) y[(size_t)b * CH + j] = nh;
    }
}

// Bahdanau attention (sparse_val==1): score=v.(common + proj), softmax over
// S, context over attender rows (SB layout). One block per batch row.
__global__ __launch_bounds__(256) void bahdanau_kernel(
    const float* __restrict__ common, const float* __restrict__ attender,
    const float* __restrict__ proj, const float* __restrict__ vvec,
    float* __restrict__ ctx, float* __restrict__ alpha_out, int D)
{
    const int b = blockIdx.x;
    const int t = threadIdx.x;
    __shared__ float part[256];
    __shared__ float sc[CS];
    __shared__ float red[CS];
    __shared__ float stat[2];
    {
        int s = t >> 1, half = t & 1;
        const float* cm = common + (size_t)(s * CB + b) * CVATT;
        const float* pj = proj + b * CVATT;
        float acc = 0.0f;
        int base = half * 256;
        for (int v = 0; v < 256; ++v)
            acc += (cm[base + v] + pj[base + v]) * vvec[base + v];
        part[t] = acc;
    }
    __syncthreads();
    if (t < CS) sc[t] = part[2 * t] + part[2 * t + 1];
    __syncthreads();
    if (t < CS) red[t] = sc[t];
    __syncthreads();
    for (int off = 64; off > 0; off >>= 1) {
        if (t < off) red[t] = fmaxf(red[t], red[t + off]);
        __syncthreads();
    }
    if (t == 0) stat[0] = red[0];
    __syncthreads();
    if (t < CS) { sc[t] = __expf(sc[t] - stat[0]); red[t] = sc[t]; }
    __syncthreads();
    for (int off = 64; off > 0; off >>= 1) {
        if (t < off) red[t] += red[t + off];
        __syncthreads();
    }
    if (t == 0) stat[1] = red[0];
    __syncthreads();
    if (t < CS) {
        sc[t] /= stat[1];
        if (alpha_out) alpha_out[b * CS + t] = sc[t];
    }
    __syncthreads();
    for (int d = t; d < D; d += 256) {
        float acc = 0.0f;
        for (int s = 0; s < CS; ++s)
            acc += sc[s] * attender[(size_t)(s * CB + b) * D + d];
        ctx[b * D + d] = acc;
    }
}

// Hierarchical triple attention + argmax triple selection. One block per b.
__global__ __launch_bounds__(256) void triple_attn_kernel(
    const float* __restrict__ all_emb, const float* __restrict__ inter,
    const float* __restrict__ a_top, float* __restrict__ c_hier,
    float* __restrict__ ksel)
{
    const int b = blockIdx.x;
    const int t = threadIdx.x;
    __shared__ float alpha[CS * CM];     // 4096
    __shared__ float rv[256];
    __shared__ int   ri[256];
    const float* ib = inter + b * CTRIP;

    for (int i = t; i < CS * CM; i += 256) {
        const float* row = all_emb + ((size_t)b * CS * CM + i) * CTRIP;
        float acc = 0.0f;
        for (int d = 0; d < CTRIP; ++d) acc += row[d] * ib[d];
        alpha[i] = acc;
    }
    __syncthreads();
    if (t < CS) {   // softmax over 32 triples per position, scaled by a_top
        int s = t;
        float mx = -3.4e38f;
        for (int m = 0; m < CM; ++m) mx = fmaxf(mx, alpha[s * CM + m]);
        float sum = 0.0f;
        for (int m = 0; m < CM; ++m) {
            float e = __expf(alpha[s * CM + m] - mx);
            alpha[s * CM + m] = e; sum += e;
        }
        float scl = a_top[b * CS + s] / sum;
        for (int m = 0; m < CM; ++m) alpha[s * CM + m] *= scl;
    }
    __syncthreads();
    // first-occurrence argmax over the 4096 attention weights
    float bv = -3.4e38f; int bi = CS * CM;
    for (int i = t; i < CS * CM; i += 256)
        if (alpha[i] > bv) { bv = alpha[i]; bi = i; }
    rv[t] = bv; ri[t] = bi;
    __syncthreads();
    for (int off = 128; off > 0; off >>= 1) {
        if (t < off) {
            if (rv[t + off] > rv[t] ||
                (rv[t + off] == rv[t] && ri[t + off] < ri[t])) {
                rv[t] = rv[t + off]; ri[t] = ri[t + off];
            }
        }
        __syncthreads();
    }
    const int best = ri[0];
    for (int d = t; d < CTRIP; d += 256) {
        float acc = 0.0f;
        const float* base = all_emb + (size_t)b * CS * CM * CTRIP + d;
        for (int i = 0; i < CS * CM; ++i) acc += alpha[i] * base[(size_t)i * CTRIP];
        c_hier[b * CTRIP + d] = acc;
        ksel[b * CTRIP + d] = all_emb[((size_t)b * CS * CM + best) * CTRIP + d];
    }
}

// x = cat(c_g, c_hier, k_sel, c_enc, wemb)
__global__ __launch_bounds__(256) void assemble_x_kernel(
    const float* __restrict__ cg, const float* __restrict__ ch,
    const float* __restrict__ ks, const float* __restrict__ ce,
    const float* __restrict__ we, float* __restrict__ x)
{
    const int b = blockIdx.x;
    for (int j = threadIdx.x; j < CDECIN; j += 256) {
        float v;
        if (j < 200)        v = cg[b * 200 + j];
        else if (j < 500)   v = ch[b * 300 + (j - 200)];
        else if (j < 800)   v = ks[b * 300 + (j - 500)];
        else if (j < 1312)  v = ce[b * 512 + (j - 800)];
        else                v = we[b * 300 + (j - 1312)];
        x[(size_t)b * CDECIN + j] = v;
    }
}

// softmax(logits) -> probs ; argmax -> wemb = word_lookup[idx]
__global__ __launch_bounds__(256) void vocab_kernel(
    const float* __restrict__ logits, const float* __restrict__ lookup,
    float* __restrict__ probs_out, float* __restrict__ wemb)
{
    const int b = blockIdx.x;
    const int t = threadIdx.x;
    __shared__ float rv[256];
    __shared__ int   ri[256];
    __shared__ float stat[2];
    const float* lg = logits + (size_t)b * CV;

    float bv = -3.4e38f; int bi = CV;
    for (int i = t; i < CV; i += 256) {
        float x = lg[i];
        if (x > bv) { bv = x; bi = i; }
    }
    rv[t] = bv; ri[t] = bi;
    __syncthreads();
    for (int off = 128; off > 0; off >>= 1) {
        if (t < off) {
            if (rv[t + off] > rv[t] ||
                (rv[t + off] == rv[t] && ri[t + off] < ri[t])) {
                rv[t] = rv[t + off]; ri[t] = ri[t + off];
            }
        }
        __syncthreads();
    }
    if (t == 0) stat[0] = rv[0];
    __syncthreads();
    const float mx = stat[0];
    float sum = 0.0f;
    for (int i = t; i < CV; i += 256) sum += __expf(lg[i] - mx);
    rv[t] = sum;
    __syncthreads();
    for (int off = 128; off > 0; off >>= 1) {
        if (t < off) rv[t] += rv[t + off];
        __syncthreads();
    }
    if (t == 0) stat[1] = rv[0];
    __syncthreads();
    const float inv = 1.0f / stat[1];
    for (int i = t; i < CV; i += 256)
        probs_out[(size_t)b * CV + i] = __expf(lg[i] - mx) * inv;
    const int best = ri[0];
    for (int d = t; d < CW; d += 256)
        wemb[b * CW + d] = lookup[(size_t)best * CW + d];
}

__global__ void zero_kernel(float* p, int n)
{
    int i = blockIdx.x * blockDim.x + threadIdx.x;
    if (i < n) p[i] = 0.0f;
}

// decoder initial hidden = encoder final hiddens (last SB rows)
__global__ __launch_bounds__(256) void init_dec_h_kernel(
    const float* __restrict__ ys0, const float* __restrict__ encO,
    float* __restrict__ h0, float* __restrict__ h1)
{
    const int b = blockIdx.x;
    for (int j = threadIdx.x; j < CH; j += 256) {
        h0[b * CH + j] = ys0 [((size_t)(CS - 1) * CB + b) * CH + j];
        h1[b * CH + j] = encO[((size_t)(CS - 1) * CB + b) * CH + j];
    }
}

// ---------------- host-side helpers ----------------
static inline int kpad(int K) { return (K + 31) & ~31; }

static inline void packB(const float* src, _Float16* dst, int N, int K,
                         int trans, hipStream_t s)
{
    long total = (long)N * kpad(K);
    pack_b_kernel<<<(unsigned)((total + 255) / 256), 256, 0, s>>>(
        src, dst, N, K, kpad(K), trans);
}

static inline void gemm(const float* A, const _Float16* Bh, float* C,
                        int M, int N, int K, hipStream_t s)
{
    dim3 block(128);
    dim3 grid((N + 63) / 64, (M + 63) / 64);
    wmma_gemm_kernel<<<grid, block, 0, s>>>(A, Bh, C, M, N, K, kpad(K));
}

static inline void zerof(float* p, int n, hipStream_t s)
{
    zero_kernel<<<(n + 255) / 256, 256, 0, s>>>(p, n);
}

extern "C" void kernel_launch(void* const* d_in, const int* in_sizes, int n_in,
                              void* d_out, int out_size, void* d_ws, size_t ws_size,
                              hipStream_t stream)
{
    (void)in_sizes; (void)n_in; (void)out_size; (void)ws_size;
    // -------- inputs (setup_inputs dict order; params dict flattened) --------
    const float* head   = (const float*)d_in[0];
    const float* tail   = (const float*)d_in[1];
    const float* rel    = (const float*)d_in[2];
    const float* wordE  = (const float*)d_in[3];
    const float* lookup = (const float*)d_in[4];
    const float* Wh_g   = (const float*)d_in[5];
    const float* Wt_g   = (const float*)d_in[6];
    const float* Wr_g   = (const float*)d_in[7];
    const float* e0_Wih = (const float*)d_in[8];
    const float* e0_Whh = (const float*)d_in[9];
    const float* e0_bih = (const float*)d_in[10];
    const float* e0_bhh = (const float*)d_in[11];
    const float* e1_Wih = (const float*)d_in[12];
    const float* e1_Whh = (const float*)d_in[13];
    const float* e1_bih = (const float*)d_in[14];
    const float* e1_bhh = (const float*)d_in[15];
    const float* Wc_enc = (const float*)d_in[16];
    const float* Wc_gr  = (const float*)d_in[17];
    const float* Wd_enc = (const float*)d_in[18];
    const float* v_enc  = (const float*)d_in[19];
    const float* Wd_gr  = (const float*)d_in[20];
    const float* v_gr   = (const float*)d_in[21];
    const float* Wd_tr  = (const float*)d_in[22];
    const float* d0_Wih = (const float*)d_in[23];
    const float* d0_Whh = (const float*)d_in[24];
    const float* d0_bih = (const float*)d_in[25];
    const float* d0_bhh = (const float*)d_in[26];
    const float* d1_Wih = (const float*)d_in[27];
    const float* d1_Whh = (const float*)d_in[28];
    const float* d1_bih = (const float*)d_in[29];
    const float* d1_bhh = (const float*)d_in[30];
    const float* Wvoc   = (const float*)d_in[31];
    float* out = (float*)d_out;                 // [T,B,V] fp32

    // -------- workspace layout; needs ~141 MB (82 MB f32 + 59 MB f16) -------
    float* W = (float*)d_ws;
    const size_t SZ_ALLEMB = (size_t)CB * CS * CM * CTRIP;    // 9,830,400
    const size_t SZ_G      = (size_t)CS * CB * (2 * CE);      //   204,800
    const size_t SZ_X      = (size_t)CS * CB * CENCIN;        //   512,000
    const size_t OFF_ALLEMB = 0;
    const size_t OFF_G      = OFF_ALLEMB + SZ_ALLEMB;
    const size_t OFF_X      = OFF_G + SZ_G;
    const size_t OFF_A      = OFF_X + SZ_X;                   // reusable region
    const size_t SZ_A       = SZ_ALLEMB;                      // 9,830,400 floats
    const size_t OFF_F16    = OFF_A + SZ_A;                   // f16 weights area

    float* all_emb = W + OFF_ALLEMB;
    float* g_buf   = W + OFF_G;
    float* Xin     = W + OFF_X;
    // region A, phase 1: graph projection scratch (dead after fuse)
    float* HH = W + OFF_A;
    float* TT = HH + (size_t)CB * CS * CM * CE;               // +3,276,800
    float* RR = TT + (size_t)CB * CS * CM * CE;
    // region A, phase 2: encoder/decoder buffers (reuse of HH/TT/RR space)
    float* Gi    = W + OFF_A;                                  // 1024*1536
    float* ys0   = Gi   + (size_t)CS * CB * C3H;               // 1024*512
    float* encO  = ys0  + (size_t)CS * CB * CH;
    float* encC  = encO + (size_t)CS * CB * CH;
    float* gC    = encC + (size_t)CS * CB * CVATT;
    float* h_enc = gC   + (size_t)CS * CB * CVATT;             // 8*512
    float* gh_b  = h_enc + CB * CH;                            // 8*1536
    float* h0    = gh_b + CB * C3H;
    float* h1    = h0 + CB * CH;
    float* pr_e  = h1 + CB * CH;                               // 8*512
    float* pr_g  = pr_e + CB * CVATT;
    float* inter = pr_g + CB * CVATT;                          // 8*300
    float* a_top = inter + CB * CTRIP;                         // 8*128
    float* c_enc = a_top + CB * CS;                            // 8*512
    float* c_g   = c_enc + CB * CH;                            // 8*200
    float* c_hi  = c_g + CB * 2 * CE;                          // 8*300
    float* ksel  = c_hi + CB * CTRIP;                          // 8*300
    float* wemb  = ksel + CB * CTRIP;                          // 8*300
    float* xdec  = wemb + CB * CW;                             // 8*1612
    float* gi0   = xdec + CB * CDECIN;                         // 8*1536
    float* gh0   = gi0 + CB * C3H;
    float* gi1   = gh0 + CB * C3H;
    float* gh1   = gi1 + CB * C3H;
    float* logit = gh1 + CB * C3H;                             // 8*40000

    // f16 packed weights [N, KP]
    _Float16* F = (_Float16*)(W + OFF_F16);
    _Float16* pWh  = F;                                        // 100 x 128
    _Float16* pWt  = pWh  + (size_t)100  * 128;
    _Float16* pWr  = pWt  + (size_t)100  * 128;
    _Float16* pE0i = pWr  + (size_t)100  * 128;                // 1536 x 512
    _Float16* pE0h = pE0i + (size_t)C3H  * 512;
    _Float16* pE1i = pE0h + (size_t)C3H  * 512;
    _Float16* pE1h = pE1i + (size_t)C3H  * 512;
    _Float16* pWcE = pE1h + (size_t)C3H  * 512;                // 512 x 512
    _Float16* pWcG = pWcE + (size_t)512  * 512;                // 512 x 224
    _Float16* pWdE = pWcG + (size_t)512  * 224;
    _Float16* pWdG = pWdE + (size_t)512  * 512;
    _Float16* pWdT = pWdG + (size_t)512  * 512;                // 300 x 512
    _Float16* pD0i = pWdT + (size_t)300  * 512;                // 1536 x 1632
    _Float16* pD0h = pD0i + (size_t)C3H  * 1632;
    _Float16* pD1i = pD0h + (size_t)C3H  * 512;
    _Float16* pD1h = pD1i + (size_t)C3H  * 512;
    _Float16* pV   = pD1h + (size_t)C3H  * 512;                // 40000 x 512

    const int BSM = CB * CS * CM;        // 32768
    const int SB  = CS * CB;             // 1024

    // ======== 0. prepack all weights to f16 [N,KP] (once per launch) ========
    packB(Wh_g,   pWh,  100,  CE,     1, stream);
    packB(Wt_g,   pWt,  100,  CE,     1, stream);
    packB(Wr_g,   pWr,  100,  CR,     1, stream);
    packB(e0_Wih, pE0i, C3H,  CENCIN, 0, stream);
    packB(e0_Whh, pE0h, C3H,  CH,     0, stream);
    packB(e1_Wih, pE1i, C3H,  CH,     0, stream);
    packB(e1_Whh, pE1h, C3H,  CH,     0, stream);
    packB(Wc_enc, pWcE, CVATT, CH,    1, stream);
    packB(Wc_gr,  pWcG, CVATT, 2*CE,  1, stream);
    packB(Wd_enc, pWdE, CVATT, CH,    1, stream);
    packB(Wd_gr,  pWdG, CVATT, CH,    1, stream);
    packB(Wd_tr,  pWdT, CTRIP, CH,    1, stream);
    packB(d0_Wih, pD0i, C3H,  CDECIN, 0, stream);
    packB(d0_Whh, pD0h, C3H,  CH,     0, stream);
    packB(d1_Wih, pD1i, C3H,  CH,     0, stream);
    packB(d1_Whh, pD1h, C3H,  CH,     0, stream);
    packB(Wvoc,   pV,   CV,   CH,     1, stream);

    // ======== 1. graph encoder attention (WMMA projections + fuse) ========
    gemm(head, pWh, HH, BSM, 100, CE, stream);
    gemm(tail, pWt, TT, BSM, 100, CE, stream);
    gemm(rel,  pWr, RR, BSM, 100, CR, stream);
    graph_fuse_kernel<<<CB * CS, 128, 0, stream>>>(HH, TT, RR, head, tail, g_buf);
    build_enc_in_kernel<<<SB, 128, 0, stream>>>(wordE, g_buf, Xin);
    {
        size_t total = SZ_ALLEMB;
        build_all_emb_kernel<<<(unsigned)((total + 255) / 256), 256, 0, stream>>>(
            head, rel, tail, all_emb);
    }

    // ======== 2. encoder GRU (2 layers): input GEMM parallel, scan serial ====
    // layer 0
    gemm(Xin, pE0i, Gi, SB, C3H, CENCIN, stream);
    zerof(h_enc, CB * CH, stream);
    for (int s = 0; s < CS; ++s) {
        gemm(h_enc, pE0h, gh_b, CB, C3H, CH, stream);
        gru_combine_kernel<<<CB, 256, 0, stream>>>(
            Gi + (size_t)s * CB * C3H, gh_b, e0_bih, e0_bhh,
            h_enc, ys0 + (size_t)s * CB * CH);
    }
    // layer 1 (input = ys0, fresh zero hidden)
    gemm(ys0, pE1i, Gi, SB, C3H, CH, stream);
    zerof(h_enc, CB * CH, stream);
    for (int s = 0; s < CS; ++s) {
        gemm(h_enc, pE1h, gh_b, CB, C3H, CH, stream);
        gru_combine_kernel<<<CB, 256, 0, stream>>>(
            Gi + (size_t)s * CB * C3H, gh_b, e1_bih, e1_bhh,
            h_enc, encO + (size_t)s * CB * CH);
    }

    // ======== 3. Bahdanau common projections (once) ========
    gemm(encO,  pWcE, encC, SB, CVATT, CH,     stream);
    gemm(g_buf, pWcG, gC,   SB, CVATT, 2 * CE, stream);

    // ======== 4. decoder init ========
    init_dec_h_kernel<<<CB, 256, 0, stream>>>(ys0, encO, h0, h1);
    zerof(wemb, CB * CW, stream);

    // ======== 5. decode loop ========
    for (int t = 0; t < CT; ++t) {
        // per-step projections of h_top (=h1, pre-update)
        gemm(h1, pWdE, pr_e,  CB, CVATT, CH, stream);
        gemm(h1, pWdG, pr_g,  CB, CVATT, CH, stream);
        gemm(h1, pWdT, inter, CB, CTRIP, CH, stream);
        // attentions
        bahdanau_kernel<<<CB, 256, 0, stream>>>(encC, encO, pr_e, v_enc,
                                                c_enc, nullptr, CH);
        bahdanau_kernel<<<CB, 256, 0, stream>>>(gC, g_buf, pr_g, v_gr,
                                                c_g, a_top, 2 * CE);
        triple_attn_kernel<<<CB, 256, 0, stream>>>(all_emb, inter, a_top,
                                                   c_hi, ksel);
        assemble_x_kernel<<<CB, 256, 0, stream>>>(c_g, c_hi, ksel, c_enc,
                                                  wemb, xdec);
        // decoder GRU cells (layer1 input is the OLD layer-0 hidden)
        gemm(xdec, pD0i, gi0, CB, C3H, CDECIN, stream);
        gemm(h0,   pD0h, gh0, CB, C3H, CH,     stream);
        gemm(h0,   pD1i, gi1, CB, C3H, CH,     stream);  // old h0
        gemm(h1,   pD1h, gh1, CB, C3H, CH,     stream);
        gru_combine_kernel<<<CB, 256, 0, stream>>>(gi0, gh0, d0_bih, d0_bhh,
                                                   h0, nullptr);
        gru_combine_kernel<<<CB, 256, 0, stream>>>(gi1, gh1, d1_bih, d1_bhh,
                                                   h1, nullptr);
        // word generator
        gemm(h1, pV, logit, CB, CV, CH, stream);
        vocab_kernel<<<CB, 256, 0, stream>>>(
            logit, lookup, out + (size_t)t * CB * CV, wemb);
    }
}
